// WindowAttention_64020782514964
// MI455X (gfx1250) — matrix operand order repaired and checked by reading, not compiled, not run on hardware
//
#include <hip/hip_runtime.h>
#include <hip/hip_bf16.h>

// ---------------- CDNA5 (gfx1250) fused Swin-V2 window attention ----------------
// One block (8 wave32s) per 64-token window; wave h owns head h end-to-end.
// All matmuls on v_wmma_f32_16x16x32_f16, f32 accumulation, LDS-staged operands.
// Shift-window mask tile is DMA'd to LDS with global_load_async_to_lds_b128
// (ASYNCcnt) at kernel entry, overlapping the QKV phase.

#define HEADS 8
#define NTOK  64
#define DIM   256
#define HD    32
#define NWIN  64
#define NBLK  4096   // B_ = 64 images * 64 windows

typedef _Float16 h16;
typedef _Float16 v16h __attribute__((ext_vector_type(16)));
typedef _Float16 v8h  __attribute__((ext_vector_type(8)));
typedef float    v8f  __attribute__((ext_vector_type(8)));

__device__ __forceinline__ v8f zero8() {
  v8f z;
  #pragma unroll
  for (int i = 0; i < 8; ++i) z[i] = 0.f;
  return z;
}

// Load 16 halves as two contiguous 8-half chunks (chunk1 at p, chunk2 at p+off2).
__device__ __forceinline__ v16h ld16(const h16* p, int off2) {
  v8h lo = *(const v8h*)(p);
  v8h hi = *(const v8h*)(p + off2);
  v16h r;
  #pragma unroll
  for (int i = 0; i < 8; ++i) { r[i] = lo[i]; r[8 + i] = hi[i]; }
  return r;
}

// A-matrix fragment (16x32 f16), source row-major (M rows, K contiguous), stride ld halves.
// Lane l<16: row r0+l, K = [k0..k0+7] then [k0+16..k0+23]; lane l+16: K offset +8.
__device__ __forceinline__ v16h load_A(const h16* m, int ld, int r0, int k0, int lane) {
  int lr = lane & 15, ks = (lane >> 4) << 3;
  return ld16(m + (size_t)(r0 + lr) * ld + k0 + ks, 16);
}

// B-matrix fragment (32x16 f16) with source stored (N,K)-row-major (i.e. B^T), stride ld.
// Lane n<16: row n0+n, K = [k0..k0+7] then [k0+16..k0+23]; lane n+16: K offset +8.
__device__ __forceinline__ v16h load_B(const h16* m, int ld, int n0, int k0, int lane) {
  int ln = lane & 15, ks = (lane >> 4) << 3;
  return ld16(m + (size_t)(n0 + ln) * ld + k0 + ks, 16);
}

__device__ __forceinline__ v8f wmma16(v16h a, v16h b, v8f c) {
  return __builtin_amdgcn_wmma_f32_16x16x32_f16(false, a, false, b, (short)0, c, false, false);
}

// ---------------- prologue kernels ----------------

__global__ void swin_cvt_weights(const float* __restrict__ qkvw, const float* __restrict__ projw,
                                 h16* __restrict__ qkvw_h, h16* __restrict__ projw_h) {
  int t = blockIdx.x * blockDim.x + threadIdx.x;
  if (t < 768 * 256) qkvw_h[t] = (h16)qkvw[t];
  int t2 = t - 768 * 256;
  if (t2 >= 0 && t2 < 256 * 256) projw_h[t2] = (h16)projw[t2];
}

// CPB MLP: (225,2) -> relu(512) -> (8).  tbl[idx*8+h]
__global__ void swin_cpb_table(const float* __restrict__ tin, const float* __restrict__ w1,
                               const float* __restrict__ b1, const float* __restrict__ w2,
                               float* __restrict__ tbl) {
  int t = blockIdx.x * blockDim.x + threadIdx.x;
  if (t >= 225 * 8) return;
  int idx = t >> 3, h = t & 7;
  float x0 = tin[idx * 2 + 0], x1 = tin[idx * 2 + 1];
  float acc = 0.f;
  for (int j = 0; j < 512; ++j) {
    float hv = fmaxf(x0 * w1[j * 2 + 0] + x1 * w1[j * 2 + 1] + b1[j], 0.f);
    acc += w2[h * 512 + j] * hv;
  }
  tbl[idx * 8 + h] = acc;
}

// biasT[h][n][m] = 16*sigmoid(tbl[rpi[n][m]][h])
__global__ void swin_cpb_gather(const float* __restrict__ tbl, const int* __restrict__ rpi,
                                float* __restrict__ biasT) {
  int t = blockIdx.x * blockDim.x + threadIdx.x;
  if (t >= HEADS * NTOK * NTOK) return;
  int h = t >> 12, nm = t & 4095;
  float v = tbl[rpi[nm] * 8 + h];
  biasT[t] = 16.f / (1.f + __expf(-v));
}

// ---------------- fused attention kernel ----------------

__launch_bounds__(256, 1)
__global__ void swin_attn_fused(const float* __restrict__ x, const float* __restrict__ mask,
                                const h16* __restrict__ qkvw, const float* __restrict__ qb,
                                const float* __restrict__ vbias, const float* __restrict__ lsc,
                                const h16* __restrict__ projw, const float* __restrict__ projb,
                                const float* __restrict__ biasT, float* __restrict__ out) {
  extern __shared__ h16 smem[];
  h16* xs   = smem;                         // 64*256 halves: x(f16) in phase1, head-out in phase3
  h16* qkvs = smem + NTOK * DIM;            // 8 heads * (q 64x32 | k 64x32 | vT 32x64)
  h16* pbuf = qkvs + HEADS * 3 * NTOK * HD; // 8 waves * 16x64 prob staging
  float* maskLds = (float*)(pbuf + HEADS * 16 * NTOK); // 64x64 f32 mask tile

  const int b    = blockIdx.x;
  const int t    = threadIdx.x;
  const int wv   = t >> 5;          // wave id == head id
  const int lane = t & 31;
  const int ln   = lane & 15;       // N index within fragment
  const int mh   = (lane >> 4) * 8; // M offset of lane half in C fragment

  // ---- phase 0: async-DMA this window's mask tile (16 KB) into LDS (ASYNCcnt) ----
  {
    const float* mw = mask + (size_t)(b & (NWIN - 1)) * NTOK * NTOK;
    #pragma unroll
    for (int i = 0; i < 4; ++i) {
      int j = t + i * 256;  // 1024 x 16-byte chunks
      unsigned lds_off = (unsigned)(size_t)(maskLds + j * 4);
      unsigned long long ga = (unsigned long long)(size_t)(mw + j * 4);
      asm volatile("global_load_async_to_lds_b128 %0, %1, off"
                   :: "v"(lds_off), "v"(ga) : "memory");
    }
  }

  // ---- phase 1: stage x tile (64x256) as f16 in LDS ----
  {
    const float4* xg = (const float4*)(x + (size_t)b * NTOK * DIM);
    #pragma unroll
    for (int i = 0; i < 16; ++i) {
      int j = t + i * 256;              // 4096 float4 per tile
      float4 f = xg[j];
      h16* d = xs + j * 4;
      d[0] = (h16)f.x; d[1] = (h16)f.y; d[2] = (h16)f.z; d[3] = (h16)f.w;
    }
  }
  __syncthreads();

  // ---- phase 2: QKV projection for this head (3 matrices of 64x32, K=256) ----
  h16* qh  = qkvs + wv * 3 * NTOK * HD;
  h16* kh  = qh + NTOK * HD;
  h16* vth = qh + 2 * NTOK * HD;        // stored transposed: vT[n][tok], ld=64
  {
    float bq[2], bv[2];
    bq[0] = qb[wv * 32 + ln];      bq[1] = qb[wv * 32 + 16 + ln];
    bv[0] = vbias[wv * 32 + ln];   bv[1] = vbias[wv * 32 + 16 + ln];
    for (int mt = 0; mt < 4; ++mt) {
      v8f acc[6];
      #pragma unroll
      for (int j = 0; j < 6; ++j) {
        float bias = (j < 2) ? bq[j] : (j >= 4 ? bv[j - 4] : 0.f);
        #pragma unroll
        for (int r = 0; r < 8; ++r) acc[j][r] = bias;
      }
      for (int kk = 0; kk < 8; ++kk) {
        v16h a = load_A(xs, DIM, mt * 16, kk * 32, lane);
        #pragma unroll
        for (int j = 0; j < 6; ++j) {
          int mat = j >> 1, nt = j & 1;
          v16h bf = load_B(qkvw, DIM, mat * DIM + wv * 32 + nt * 16, kk * 32, lane);
          acc[j] = wmma16(a, bf, acc[j]);
        }
      }
      #pragma unroll
      for (int j = 0; j < 6; ++j) {
        int mat = j >> 1, nt = j & 1;
        #pragma unroll
        for (int r = 0; r < 8; ++r) {
          int mrow = mt * 16 + mh + r, ncol = nt * 16 + ln;
          if (mat == 0)      qh[mrow * HD + ncol] = (h16)acc[j][r];
          else if (mat == 1) kh[mrow * HD + ncol] = (h16)acc[j][r];
          else               vth[ncol * NTOK + mrow] = (h16)acc[j][r];
        }
      }
    }
  }

  // ---- phase 2b: cosine-attn normalization; fold exp(min(logit_scale,log100)) into q ----
  {
    float lscale = __expf(fminf(lsc[wv], 4.6051702f));
    for (int rr = lane; rr < NTOK; rr += 32) {
      h16* qr = qh + rr * HD;
      h16* kr = kh + rr * HD;
      float sq = 0.f, sk = 0.f;
      #pragma unroll
      for (int c = 0; c < HD; ++c) {
        float fq = (float)qr[c], fk = (float)kr[c];
        sq += fq * fq; sk += fk * fk;
      }
      float iq = lscale / fmaxf(sqrtf(sq), 1e-12f);
      float ik = 1.f    / fmaxf(sqrtf(sk), 1e-12f);
      #pragma unroll
      for (int c = 0; c < HD; ++c) {
        qr[c] = (h16)((float)qr[c] * iq);
        kr[c] = (h16)((float)kr[c] * ik);
      }
    }
  }
  // all async mask DMA for this wave must land before the block-wide barrier
  asm volatile("s_wait_asynccnt 0x0" ::: "memory");
  __syncthreads();   // QKV + mask DMA done; xs may now be reused as head-out

  // ---- phase 3: attention, softmax, P@V (16 query rows at a time) ----
  {
    const float* biasH = biasT + (size_t)wv * NTOK * NTOK;
    h16* pw = pbuf + wv * 16 * NTOK;

    for (int mt = 0; mt < 4; ++mt) {
      // scores: (16x32) @ (32x64)^T -> 4 C tiles, single K step (K=32)
      v8f s[4];
      v16h aq = load_A(qh, HD, mt * 16, 0, lane);
      #pragma unroll
      for (int nt = 0; nt < 4; ++nt) {
        v16h bk = load_B(kh, HD, nt * 16, 0, lane);
        s[nt] = wmma16(aq, bk, zero8());
      }
      // + CPB bias (global, L2-resident) + shift-window mask (LDS)
      #pragma unroll
      for (int nt = 0; nt < 4; ++nt) {
        int ncol = nt * 16 + ln;
        #pragma unroll
        for (int r = 0; r < 8; ++r) {
          int mrow = mt * 16 + mh + r;
          s[nt][r] += biasH[mrow * NTOK + ncol] + maskLds[mrow * NTOK + ncol];
        }
      }
      // row softmax: reduce across the 16-lane half (xor 1,2,4,8 keeps halves separate)
      #pragma unroll
      for (int r = 0; r < 8; ++r) {
        float mx = fmaxf(fmaxf(s[0][r], s[1][r]), fmaxf(s[2][r], s[3][r]));
        #pragma unroll
        for (int off = 1; off < 16; off <<= 1) mx = fmaxf(mx, __shfl_xor(mx, off, 32));
        float sum = 0.f;
        #pragma unroll
        for (int nt = 0; nt < 4; ++nt) { float e = __expf(s[nt][r] - mx); s[nt][r] = e; sum += e; }
        #pragma unroll
        for (int off = 1; off < 16; off <<= 1) sum += __shfl_xor(sum, off, 32);
        float inv = 1.f / sum;
        #pragma unroll
        for (int nt = 0; nt < 4; ++nt) s[nt][r] *= inv;
      }
      // stage probs as f16 A-source (16x64, row-major)
      #pragma unroll
      for (int nt = 0; nt < 4; ++nt)
        #pragma unroll
        for (int r = 0; r < 8; ++r)
          pw[(mh + r) * NTOK + nt * 16 + ln] = (h16)s[nt][r];

      // P(16x64) @ V(64x32): K in 2 steps, vT is (N,K)-row-major
      v8f o[2] = { zero8(), zero8() };
      #pragma unroll
      for (int kk = 0; kk < 2; ++kk) {
        v16h pa = load_A(pw, NTOK, 0, kk * 32, lane);
        #pragma unroll
        for (int nt2 = 0; nt2 < 2; ++nt2) {
          v16h bv = load_B(vth, NTOK, nt2 * 16, kk * 32, lane);
          o[nt2] = wmma16(pa, bv, o[nt2]);
        }
      }
      // head output -> xs (reused), columns [h*32, h*32+32)
      #pragma unroll
      for (int nt2 = 0; nt2 < 2; ++nt2)
        #pragma unroll
        for (int r = 0; r < 8; ++r)
          xs[(size_t)(mt * 16 + mh + r) * DIM + wv * 32 + nt2 * 16 + ln] = (h16)o[nt2][r];
    }
  }
  __syncthreads();

  // ---- phase 4: output projection (64x256) @ (256x256)^T + b, this wave's 32 cols ----
  {
    float pb2[2] = { projb[wv * 32 + ln], projb[wv * 32 + 16 + ln] };
    float* og = out + (size_t)b * NTOK * DIM;
    for (int mt = 0; mt < 4; ++mt) {
      v8f acc[2];
      #pragma unroll
      for (int nt = 0; nt < 2; ++nt)
        #pragma unroll
        for (int r = 0; r < 8; ++r) acc[nt][r] = pb2[nt];
      for (int kk = 0; kk < 8; ++kk) {
        v16h a = load_A(xs, DIM, mt * 16, kk * 32, lane);
        #pragma unroll
        for (int nt = 0; nt < 2; ++nt) {
          v16h bf = load_B(projw, DIM, wv * 32 + nt * 16, kk * 32, lane);
          acc[nt] = wmma16(a, bf, acc[nt]);
        }
      }
      #pragma unroll
      for (int nt = 0; nt < 2; ++nt)
        #pragma unroll
        for (int r = 0; r < 8; ++r)
          og[(size_t)(mt * 16 + mh + r) * DIM + wv * 32 + nt * 16 + ln] = acc[nt][r];
    }
  }
}

// ---------------- host launch ----------------

extern "C" void kernel_launch(void* const* d_in, const int* in_sizes, int n_in,
                              void* d_out, int out_size, void* d_ws, size_t ws_size,
                              hipStream_t stream) {
  const float* x      = (const float*)d_in[0];
  const float* mask   = (const float*)d_in[1];
  const float* qkv_w  = (const float*)d_in[2];
  const float* q_bias = (const float*)d_in[3];
  const float* v_bias = (const float*)d_in[4];
  const float* lscale = (const float*)d_in[5];
  const float* cpb_w1 = (const float*)d_in[6];
  const float* cpb_b1 = (const float*)d_in[7];
  const float* cpb_w2 = (const float*)d_in[8];
  const float* proj_w = (const float*)d_in[9];
  const float* proj_b = (const float*)d_in[10];
  const float* rct    = (const float*)d_in[11];
  const int*   rpi    = (const int*)d_in[12];
  float* out = (float*)d_out;

  // workspace layout
  char* ws = (char*)d_ws;
  h16*  qkvw_h  = (h16*)ws;                                     // 768*256 f16
  h16*  projw_h = (h16*)(ws + 768 * 256 * 2);                   // 256*256 f16
  float* tbl    = (float*)(ws + 768 * 256 * 2 + 256 * 256 * 2);            // 225*8 f32 (pad 2048)
  float* biasT  = (float*)(ws + 768 * 256 * 2 + 256 * 256 * 2 + 2048 * 4); // 8*64*64 f32

  swin_cvt_weights<<<(768 * 256 + 256 * 256) / 256, 256, 0, stream>>>(qkv_w, proj_w, qkvw_h, projw_h);
  swin_cpb_table<<<(225 * 8 + 255) / 256, 256, 0, stream>>>(rct, cpb_w1, cpb_b1, cpb_w2, tbl);
  swin_cpb_gather<<<(HEADS * NTOK * NTOK) / 256, 256, 0, stream>>>(tbl, rpi, biasT);

  const size_t shmem = (size_t)(NTOK * DIM + HEADS * 3 * NTOK * HD + HEADS * 16 * NTOK) * sizeof(h16)
                     + (size_t)NTOK * NTOK * sizeof(float); // 160 KB
  (void)hipFuncSetAttribute(reinterpret_cast<const void*>(swin_attn_fused),
                            hipFuncAttributeMaxDynamicSharedMemorySize, (int)shmem);
  swin_attn_fused<<<NBLK, 256, shmem, stream>>>(x, mask, qkvw_h, q_bias, v_bias, lscale,
                                                projw_h, proj_b, biasT, out);
}
